// GraphTransformerLayer_81793357185279
// MI455X (gfx1250) — matrix operand order repaired and checked by reading, not compiled, not run on hardware
//
#include <hip/hip_runtime.h>

// ---------------------------------------------------------------------------
// Types for CDNA5 WMMA (wave32): v_wmma_f32_16x16x32_bf16
// ---------------------------------------------------------------------------
typedef __attribute__((ext_vector_type(16))) __bf16          v16bf;
typedef __attribute__((ext_vector_type(8)))  float           v8f;
typedef __attribute__((ext_vector_type(8)))  unsigned short  u16x8;
typedef __attribute__((ext_vector_type(16))) unsigned short  u16x16;

// Problem constants (from reference setup_inputs)
#define BB   8
#define DD   64
#define NN   1024
#define LL   12
#define DL   768          // D*L
#define SS   2
#define NHD  2
#define EE   128          // 2*C_OUT
#define HH   13
#define COUT 64
#define NEGV (-9.0e15f)

__device__ __forceinline__ unsigned short f32_to_bf16(float f) {
  unsigned int u = __float_as_uint(f);
  u += 0x7FFFu + ((u >> 16) & 1u);   // round-to-nearest-even
  return (unsigned short)(u >> 16);
}
__device__ __forceinline__ float bf16_to_f32(unsigned short h) {
  return __uint_as_float(((unsigned int)h) << 16);
}

__device__ __forceinline__ v16bf load_frag(const unsigned short* p0,
                                           const unsigned short* p1) {
  u16x8 lo = *(const u16x8*)p0;       // 16B global_load_b128
  u16x8 hi = *(const u16x8*)p1;       // 16B global_load_b128
  u16x16 u = __builtin_shufflevector(lo, hi, 0,1,2,3,4,5,6,7,
                                             8,9,10,11,12,13,14,15);
  union { u16x16 u; v16bf b; } cvt; cvt.u = u;
  return cvt.b;
}

// ---------------------------------------------------------------------------
// Generic bf16 GEMM:  C[M x Nc] = A[M x K] * Bt[Nc x K]^T   (f32 accumulate)
//   A  : row-major, K contiguous (lda = K)
//   Bt : B stored transposed, K contiguous (ldb = K)
//   MODE 0: store bf16        MODE 1: + bias[col], store bf16
//   MODE 2: ReLU, store f32
// Block = 128 threads (4 waves); block tile 64x64; wave tile 16x64.
// ---------------------------------------------------------------------------
template<int MODE>
__global__ __launch_bounds__(128) void gemm_bf16_wmma(
    const unsigned short* __restrict__ A,
    const unsigned short* __restrict__ Bt,
    void* __restrict__ Cp,
    const float* __restrict__ bias,
    int M, int Nc, int K,
    long long sA, long long sB, long long sC)
{
  const int lane = threadIdx.x & 31;
  const int wave = threadIdx.x >> 5;
  const int hl   = lane >> 4;        // which 16-lane half
  const int l16  = lane & 15;
  const int m0   = blockIdx.y * 64 + wave * 16;
  const int n0   = blockIdx.x * 64;
  const long long zb = blockIdx.z;

  const unsigned short* Ab  = A  + zb * sA;
  const unsigned short* Btb = Bt + zb * sB;

  v8f acc[4] = {};

  // per-lane A row (A fragment lane layout: row = m0 + l16)
  const unsigned short* arow = Ab + (long long)(m0 + l16) * K;

  for (int k0 = 0; k0 < K; k0 += 32) {
    // A 16x32 bf16 fragment: elems 0..7 = K k0+hl*8.., elems 8..15 = K k0+16+hl*8..
    v16bf af = load_frag(arow + k0 + hl * 8, arow + k0 + 16 + hl * 8);
    __builtin_prefetch(arow + k0 + 32, 0, 0);
#pragma unroll
    for (int c = 0; c < 4; ++c) {
      // B 32x16 bf16 fragment: col = l16, elems 0..7 = K k0+hl*8.., elems 8..15 = K k0+16+hl*8..
      const unsigned short* brow =
          Btb + (long long)(n0 + c * 16 + l16) * K + k0 + hl * 8;
      v16bf bf = load_frag(brow, brow + 16);
      acc[c] = __builtin_amdgcn_wmma_f32_16x16x32_bf16(
          false, af, false, bf, (short)0, acc[c], false, false);
    }
  }

  // C/D layout: VGPR r -> row m0 + r + 8*hl, col = n0 + c*16 + l16
#pragma unroll
  for (int c = 0; c < 4; ++c) {
    const int col = n0 + c * 16 + l16;
    float bv = 0.f;
    if (MODE == 1) bv = bias[col];
#pragma unroll
    for (int r = 0; r < 8; ++r) {
      const int row = m0 + r + 8 * hl;
      float v = acc[c][r];
      if (MODE == 2) {
        v = v > 0.f ? v : 0.f;
        ((float*)Cp)[zb * sC + (long long)row * Nc + col] = v;
      } else {
        v += bv;
        ((unsigned short*)Cp)[zb * sC + (long long)row * Nc + col] = f32_to_bf16(v);
      }
    }
  }
}

// ---------------------------------------------------------------------------
// Packing / conversion kernels
// ---------------------------------------------------------------------------
// Xp[b][d*12+l][n] = bf16( x[b][d][n][l] )   (A operand of conv GEMMs; out[0])
__global__ void pack_x_kernel(const float* __restrict__ x,
                              unsigned short* __restrict__ Xp) {
  const long long total = (long long)BB * DL * NN;
  for (long long idx = (long long)blockIdx.x * blockDim.x + threadIdx.x;
       idx < total; idx += (long long)gridDim.x * blockDim.x) {
    int n  = (int)(idx % NN);
    int dl = (int)((idx / NN) % DL);
    int b  = (int)(idx / ((long long)NN * DL));
    int d = dl / LL, l = dl % LL;
    Xp[idx] = f32_to_bf16(x[(((long long)b * DD + d) * NN + n) * LL + l]);
  }
}

// Kp[b][n][dl] = bf16(x + nv1[n][d]);  Qp[b][n][dl] = bf16(x + nv2[d][n])
__global__ void pack_kq_kernel(const float* __restrict__ x,
                               const float* __restrict__ nv1,
                               const float* __restrict__ nv2,
                               unsigned short* __restrict__ Kp,
                               unsigned short* __restrict__ Qp) {
  const long long total = (long long)BB * NN * DL;
  for (long long idx = (long long)blockIdx.x * blockDim.x + threadIdx.x;
       idx < total; idx += (long long)gridDim.x * blockDim.x) {
    int dl = (int)(idx % DL);
    int n  = (int)((idx / DL) % NN);
    int b  = (int)(idx / ((long long)DL * NN));
    int d = dl / LL, l = dl % LL;
    float xv = x[(((long long)b * DD + d) * NN + n) * LL + l];
    Kp[idx] = f32_to_bf16(xv + nv1[n * DD + d]);
    Qp[idx] = f32_to_bf16(xv + nv2[d * NN + n]);
  }
}

__global__ void cvt_bf16_kernel(const float* __restrict__ src,
                                unsigned short* __restrict__ dst, long long n) {
  for (long long i = (long long)blockIdx.x * blockDim.x + threadIdx.x;
       i < n; i += (long long)gridDim.x * blockDim.x)
    dst[i] = f32_to_bf16(src[i]);
}

// abf[s][w][v] = bf16( support[s][v][w] )  (transposed: conv B operand + mask)
__global__ void pack_adj_kernel(const float* __restrict__ sup,
                                unsigned short* __restrict__ abf) {
  const long long total = (long long)SS * NN * NN;
  for (long long idx = (long long)blockIdx.x * blockDim.x + threadIdx.x;
       idx < total; idx += (long long)gridDim.x * blockDim.x) {
    int v = (int)(idx % NN);
    int w = (int)((idx / NN) % NN);
    int s = (int)(idx / ((long long)NN * NN));
    abf[idx] = f32_to_bf16(sup[((long long)s * NN + v) * NN + w]);
  }
}

// ---------------------------------------------------------------------------
// Masked column-softmax (done per row of eT): att[b][w][v]
// eT[b][w][v] holds relu(e)[v,w]; mask where a[v][w]==0 -> NEG
// ---------------------------------------------------------------------------
__global__ __launch_bounds__(256) void softmax_mask(
    const float* __restrict__ eT, const unsigned short* __restrict__ abf,
    unsigned short* __restrict__ att) {
  __shared__ float red[256];
  const int row = blockIdx.x;               // b*N + w
  const int w   = row % NN;
  const float* erow = eT + (long long)row * NN;
  const unsigned short* mrow = abf + (long long)w * NN;
  unsigned short* orow = att + (long long)row * NN;
  const int tid = threadIdx.x;

  float mx = -3.0e38f;
  for (int v = tid; v < NN; v += 256) {
    float e = (mrow[v] != 0) ? erow[v] : NEGV;
    mx = fmaxf(mx, e);
  }
  red[tid] = mx; __syncthreads();
  for (int s = 128; s > 0; s >>= 1) {
    if (tid < s) red[tid] = fmaxf(red[tid], red[tid + s]);
    __syncthreads();
  }
  mx = red[0]; __syncthreads();

  float sum = 0.f;
  for (int v = tid; v < NN; v += 256) {
    float e = (mrow[v] != 0) ? erow[v] : NEGV;
    sum += __expf(e - mx);
  }
  red[tid] = sum; __syncthreads();
  for (int s = 128; s > 0; s >>= 1) {
    if (tid < s) red[tid] += red[tid + s];
    __syncthreads();
  }
  const float inv = 1.f / red[0];
  for (int v = tid; v < NN; v += 256) {
    float e = (mrow[v] != 0) ? erow[v] : NEGV;
    orow[v] = f32_to_bf16(__expf(e - mx) * inv);
  }
}

// ---------------------------------------------------------------------------
// feats[b][i][d] = mean over (l, n) of outs[i][b][d*12+l][n]
// ---------------------------------------------------------------------------
__global__ __launch_bounds__(128) void means_kernel(
    const unsigned short* __restrict__ outs, float* __restrict__ feats) {
  __shared__ float red[128];
  const int idx = blockIdx.x;               // i*(B*D) + b*D + d
  const int d = idx % DD;
  const int b = (idx / DD) % BB;
  const int i = idx / (DD * BB);
  const unsigned short* base =
      outs + (((long long)i * BB + b) * DL + (long long)d * LL) * NN;
  float s = 0.f;
  for (int t = threadIdx.x; t < LL * NN; t += 128) {
    int l = t >> 10, n = t & (NN - 1);
    s += bf16_to_f32(base[(long long)l * NN + n]);
  }
  red[threadIdx.x] = s; __syncthreads();
  for (int st = 64; st > 0; st >>= 1) {
    if (threadIdx.x < st) red[threadIdx.x] += red[threadIdx.x + st];
    __syncthreads();
  }
  if (threadIdx.x == 0)
    feats[((long long)b * HH + i) * DD + d] = red[0] / (float)(LL * NN);
}

// logits[b][j] = feats_flat[b] . attn_w[j] + attn_b[j]; gate = softmax(logits)
__global__ __launch_bounds__(128) void gate_kernel(
    const float* __restrict__ feats, const float* __restrict__ attn_w,
    const float* __restrict__ attn_b, float* __restrict__ gate) {
  __shared__ float red[128];
  __shared__ float logits[HH];
  const int b = blockIdx.x;
  const float* f = feats + (long long)b * HH * DD;
  for (int j = 0; j < HH; ++j) {
    float p = 0.f;
    for (int k = threadIdx.x; k < HH * DD; k += 128)
      p += f[k] * attn_w[j * (HH * DD) + k];
    red[threadIdx.x] = p; __syncthreads();
    for (int st = 64; st > 0; st >>= 1) {
      if (threadIdx.x < st) red[threadIdx.x] += red[threadIdx.x + st];
      __syncthreads();
    }
    if (threadIdx.x == 0) logits[j] = red[0] + attn_b[j];
    __syncthreads();
  }
  if (threadIdx.x == 0) {
    float mx = logits[0];
    for (int j = 1; j < HH; ++j) mx = fmaxf(mx, logits[j]);
    float s = 0.f, ex[HH];
    for (int j = 0; j < HH; ++j) { ex[j] = __expf(logits[j] - mx); s += ex[j]; }
    for (int j = 0; j < HH; ++j) gate[b * HH + j] = ex[j] / s;
  }
}

// out[b][o][n][l] = sum_d final_w[o][d] * (sum_i gate[b][i]*outs[i][b][d*12+l][n]) + final_b[o]
__global__ __launch_bounds__(256) void final_kernel(
    const unsigned short* __restrict__ outs, const float* __restrict__ gate,
    const float* __restrict__ fw, const float* __restrict__ fb,
    float* __restrict__ out) {
  __shared__ float h[DL];
  __shared__ float g[HH];
  const int n = blockIdx.x;
  const int b = blockIdx.y;
  if (threadIdx.x < HH) g[threadIdx.x] = gate[b * HH + threadIdx.x];
  __syncthreads();
  for (int dl = threadIdx.x; dl < DL; dl += 256) {
    float s = 0.f;
#pragma unroll
    for (int i = 0; i < HH; ++i)
      s += g[i] * bf16_to_f32(outs[(((long long)i * BB + b) * DL + dl) * NN + n]);
    h[dl] = s;
  }
  __syncthreads();
  for (int t = threadIdx.x; t < COUT * LL; t += 256) {
    int o = t / LL, l = t % LL;
    float s = fb[o];
    for (int d = 0; d < DD; ++d) s += fw[o * DD + d] * h[d * LL + l];
    out[(((long long)b * COUT + o) * NN + n) * LL + l] = s;
  }
}

// ---------------------------------------------------------------------------
// Host orchestration
// ---------------------------------------------------------------------------
extern "C" void kernel_launch(void* const* d_in, const int* in_sizes, int n_in,
                              void* d_out, int out_size, void* d_ws, size_t ws_size,
                              hipStream_t stream) {
  (void)in_sizes; (void)n_in; (void)out_size; (void)ws_size;
  const float* x       = (const float*)d_in[0];
  const float* support = (const float*)d_in[1];
  const float* nv1     = (const float*)d_in[2];
  const float* nv2     = (const float*)d_in[3];
  const float* key_w   = (const float*)d_in[4];
  const float* key_b   = (const float*)d_in[5];
  const float* query_w = (const float*)d_in[6];
  const float* query_b = (const float*)d_in[7];
  const float* attn_w  = (const float*)d_in[8];
  const float* attn_b  = (const float*)d_in[9];
  const float* final_w = (const float*)d_in[10];
  const float* final_b = (const float*)d_in[11];
  float* out = (float*)d_out;

  char* ws = (char*)d_ws;
  size_t off = 0;
  auto alloc = [&](size_t bytes) -> size_t {
    off = (off + 255) & ~(size_t)255;
    size_t o = off; off += bytes; return o;
  };
  const long long SL = (long long)BB * DL * NN;            // out-slice elems
  unsigned short* outs    = (unsigned short*)(ws + alloc((size_t)HH * SL * 2));
  unsigned short* Kp      = (unsigned short*)(ws + alloc((size_t)BB * NN * DL * 2));
  unsigned short* Qp      = (unsigned short*)(ws + alloc((size_t)BB * NN * DL * 2));
  unsigned short* kwbf    = (unsigned short*)(ws + alloc((size_t)NHD * EE * DL * 2));
  unsigned short* qwbf    = (unsigned short*)(ws + alloc((size_t)NHD * EE * DL * 2));
  unsigned short* abf     = (unsigned short*)(ws + alloc((size_t)SS * NN * NN * 2));
  unsigned short* keysbf  = (unsigned short*)(ws + alloc((size_t)NHD * BB * NN * EE * 2));
  unsigned short* querybf = (unsigned short*)(ws + alloc((size_t)NHD * BB * NN * EE * 2));
  float*          eT      = (float*)(ws + alloc((size_t)BB * NN * NN * 4));
  unsigned short* attbf   = (unsigned short*)(ws + alloc((size_t)BB * NN * NN * 2));
  float*          feats   = (float*)(ws + alloc((size_t)BB * HH * DD * 4));
  float*          gate    = (float*)(ws + alloc((size_t)BB * HH * 4));

  // ---- pack / convert ----
  pack_x_kernel <<<4096, 256, 0, stream>>>(x, outs);   // out[0] = x (bf16 packed)
  pack_kq_kernel<<<4096, 256, 0, stream>>>(x, nv1, nv2, Kp, Qp);
  cvt_bf16_kernel<<<256, 256, 0, stream>>>(key_w,   kwbf, (long long)NHD * EE * DL);
  cvt_bf16_kernel<<<256, 256, 0, stream>>>(query_w, qwbf, (long long)NHD * EE * DL);
  pack_adj_kernel<<<4096, 256, 0, stream>>>(support, abf);

  const long long sX  = (long long)DL * NN;   // per-batch A/C stride (convs)
  const long long sKQ = (long long)NN * DL;   // per-batch stride of Kp/Qp
  const long long sKE = (long long)NN * EE;   // per-batch stride of keys/query bf16
  const long long sNN = (long long)NN * NN;   // per-batch stride of eT / attbf

  // ---- adjacency graph convolutions (ih == 0) ----
  for (int s = 0; s < SS; ++s) {
    const int base = 1 + s * 6;
    dim3 gg(NN / 64, DL / 64, BB);
    gemm_bf16_wmma<0><<<gg, 128, 0, stream>>>(
        outs, abf + (long long)s * sNN, outs + (long long)base * SL,
        nullptr, DL, NN, NN, sX, 0LL, sX);
    gemm_bf16_wmma<0><<<gg, 128, 0, stream>>>(
        outs + (long long)base * SL, abf + (long long)s * sNN,
        outs + (long long)(base + 1) * SL, nullptr, DL, NN, NN, sX, 0LL, sX);
  }

  // ---- attention heads ----
  for (int h = 0; h < NHD; ++h) {
    dim3 gk(EE / 64, NN / 64, BB);
    // keys[b][n][e], query[b][m][e]  (bf16 with bias)
    gemm_bf16_wmma<1><<<gk, 128, 0, stream>>>(
        Kp, kwbf + (long long)h * EE * DL, keysbf + (long long)h * BB * sKE,
        key_b + h * EE, NN, EE, DL, sKQ, 0LL, sKE);
    gemm_bf16_wmma<1><<<gk, 128, 0, stream>>>(
        Qp, qwbf + (long long)h * EE * DL, querybf + (long long)h * BB * sKE,
        query_b + h * EE, NN, EE, DL, sKQ, 0LL, sKE);
    // eT[b][w][v] = relu( sum_e query[e,w] * keys[e,v] )   (f32)
    dim3 ge(NN / 64, NN / 64, BB);
    gemm_bf16_wmma<2><<<ge, 128, 0, stream>>>(
        querybf + (long long)h * BB * sKE, keysbf + (long long)h * BB * sKE,
        eT, nullptr, NN, NN, EE, sKE, sKE, sNN);
    for (int s = 0; s < SS; ++s) {
      softmax_mask<<<BB * NN, 256, 0, stream>>>(eT, abf + (long long)s * sNN, attbf);
      const int base = 1 + s * 6 + 2 * (h + 1);
      dim3 gg(NN / 64, DL / 64, BB);
      gemm_bf16_wmma<0><<<gg, 128, 0, stream>>>(
          outs, attbf, outs + (long long)base * SL,
          nullptr, DL, NN, NN, sX, sNN, sX);
      gemm_bf16_wmma<0><<<gg, 128, 0, stream>>>(
          outs + (long long)base * SL, attbf, outs + (long long)(base + 1) * SL,
          nullptr, DL, NN, NN, sX, sNN, sX);
    }
  }

  // ---- gating + final projection ----
  means_kernel<<<HH * BB * DD, 128, 0, stream>>>(outs, feats);
  gate_kernel <<<BB, 128, 0, stream>>>(feats, attn_w, attn_b, gate);
  dim3 gf(NN, BB);
  final_kernel<<<gf, 256, 0, stream>>>(outs, gate, final_w, final_b, out);
}
